// MultiHeadGATLayer_1245540516471
// MI455X (gfx1250) — matrix operand order repaired and checked by reading, not compiled, not run on hardware
//
#include <hip/hip_runtime.h>
#include <hip/hip_bf16.h>

#define IN_DIM 256
#define DHEAD  32
#define NHEAD  8
#define NEG_SLOPE 0.01f

typedef __attribute__((ext_vector_type(16))) _Float16 v16h;
typedef __attribute__((ext_vector_type(8)))  _Float16 v8h;
typedef __attribute__((ext_vector_type(8)))  float    v8f;

// ---------------- conversion kernels ----------------

__global__ void conv_h_f16(const float* __restrict__ h, _Float16* __restrict__ h16, int total) {
    int i = blockIdx.x * blockDim.x + threadIdx.x;
    if (i < total) h16[i] = (_Float16)h[i];
}

// Pack W (H, 256, 32) fp32 into f16 B-fragments.
// Fragment layout per 32x16 (KxN) tile: lane 0-15 -> N=lane, K=kb*32+{0..7,16..23} (j order);
// lane 16-31 -> N=lane-16, K=kb*32+{8..15,24..31}. Buffer index:
//   (((head*8+kb)*2+nt)*32 + lane)*16 + j
__global__ void pack_W_f16(const float* __restrict__ W, _Float16* __restrict__ Bp) {
    int idx = blockIdx.x * blockDim.x + threadIdx.x;
    if (idx >= NHEAD * 8 * 2 * 32 * 16) return;
    int j    =  idx        & 15;
    int lane = (idx >> 4)  & 31;
    int nt   = (idx >> 9)  & 1;
    int kb   = (idx >> 10) & 7;
    int head = (idx >> 13) & 7;
    int k = kb * 32 + ((j & 8) ? 16 : 0) + ((lane & 16) ? 8 : 0) + (j & 7);
    int d = nt * 16 + (lane & 15);
    Bp[idx] = (_Float16)W[((size_t)head * IN_DIM + k) * DHEAD + d];
}

// ---------------- WMMA GEMM: z = h @ W ----------------
// Block = 256 threads = 8 waves. Wave w computes head w for a 16-node row strip.
__global__ __launch_bounds__(256) void gemm_z(const _Float16* __restrict__ h16,
                                              const _Float16* __restrict__ Bp,
                                              float* __restrict__ z, int n) {
    const int lane = threadIdx.x & 31;
    const int head = threadIdx.x >> 5;
    const int m0   = blockIdx.x * 16;
    const int row  = lane & 15;
    const int klo  = (lane < 16) ? 0 : 8;   // A-fragment K sub-group per lane half

    int r = m0 + row; if (r >= n) r = n - 1;           // clamp tail loads
    const _Float16* arow = h16 + (size_t)r * IN_DIM;

    v8f acc0 = {}; v8f acc1 = {};
    #pragma unroll
    for (int kb = 0; kb < 8; ++kb) {
        // A fragment (16x32 f16): halves 0-7 = K[kb*32+klo .. +7], 8-15 = K[kb*32+16+klo .. +7]
        const _Float16* ap = arow + kb * 32;
        v8h alo = *(const v8h*)(ap + klo);
        v8h ahi = *(const v8h*)(ap + 16 + klo);
        v16h afrag;
        #pragma unroll
        for (int i = 0; i < 8; ++i) { afrag[i] = alo[i]; afrag[8 + i] = ahi[i]; }

        // B fragments from packed buffer (one contiguous 32B load per lane)
        const _Float16* bbase = Bp + ((((size_t)head * 8 + kb) * 2) * 32 + lane) * 16;
        v16h b0 = *(const v16h*)(bbase);
        v16h b1 = *(const v16h*)(bbase + 32 * 16);

        acc0 = __builtin_amdgcn_wmma_f32_16x16x32_f16(false, afrag, false, b0,
                                                      (short)0, acc0, false, false);
        acc1 = __builtin_amdgcn_wmma_f32_16x16x32_f16(false, afrag, false, b1,
                                                      (short)0, acc1, false, false);
    }

    // C/D layout: VGPR v: lanes 0-15 -> M=v, N=lane; lanes 16-31 -> M=v+8, N=lane-16
    const int col  = lane & 15;
    const int mofs = (lane < 16) ? 0 : 8;
    #pragma unroll
    for (int v = 0; v < 8; ++v) {
        int node = m0 + mofs + v;
        if (node < n) {
            z[(size_t)node * IN_DIM + head * DHEAD + col]      = acc0[v];
            z[(size_t)node * IN_DIM + head * DHEAD + 16 + col] = acc1[v];
        }
    }
}

// ---------------- attention scores el/er ----------------
__global__ void attn_scores(const float* __restrict__ z, const float* __restrict__ attn_w,
                            float* __restrict__ el, float* __restrict__ er, int nh_total) {
    int idx = blockIdx.x * blockDim.x + threadIdx.x;   // idx = n*8 + head
    if (idx >= nh_total) return;
    int head = idx & 7;
    const float4* zr = (const float4*)(z + (size_t)idx * DHEAD);
    const float4* al = (const float4*)(attn_w + head * 2 * DHEAD);
    const float4* ar = (const float4*)(attn_w + head * 2 * DHEAD + DHEAD);
    float sl = 0.f, sr = 0.f;
    #pragma unroll
    for (int q = 0; q < 8; ++q) {
        float4 zv = zr[q], av = al[q], bv = ar[q];
        sl += zv.x * av.x + zv.y * av.y + zv.z * av.z + zv.w * av.w;
        sr += zv.x * bv.x + zv.y * bv.y + zv.z * bv.z + zv.w * bv.w;
    }
    el[idx] = sl; er[idx] = sr;
}

// ---------------- init: zero out, m_enc, den ----------------
__global__ void init_bufs(float* __restrict__ out, unsigned* __restrict__ m_enc,
                          float* __restrict__ den, int out_total, int nh_total) {
    int i = blockIdx.x * blockDim.x + threadIdx.x;
    if (i < out_total) out[i] = 0.f;
    if (i < nh_total) { m_enc[i] = 0u; den[i] = 0.f; }   // enc 0 decodes to -NaN => below everything
}

// ---------------- edge pass 1: leaky-relu + segment max ----------------
__global__ void edge_max(const float* __restrict__ el, const float* __restrict__ er,
                         const int* __restrict__ src, const int* __restrict__ dst,
                         float* __restrict__ ew, unsigned* __restrict__ m_enc, int e_total) {
    int t = blockIdx.x * blockDim.x + threadIdx.x;      // t = eid*8 + head
    int eid = t >> 3, head = t & 7;
    if (eid >= e_total) return;
    int s = src[eid], d0 = dst[eid];
    float v = el[s * NHEAD + head] + er[d0 * NHEAD + head];
    v = (v > 0.f) ? v : v * NEG_SLOPE;
    ew[t] = v;
    unsigned bits = __float_as_uint(v);
    unsigned key  = (bits & 0x80000000u) ? ~bits : (bits | 0x80000000u);  // order-preserving
    atomicMax(m_enc + d0 * NHEAD + head, key);
}

// ---------------- decode m (in place), non-finite -> 0 ----------------
__global__ void decode_m(unsigned* __restrict__ m_enc, int nh_total) {
    int i = blockIdx.x * blockDim.x + threadIdx.x;
    if (i >= nh_total) return;
    unsigned key  = m_enc[i];
    unsigned bits = (key & 0x80000000u) ? (key & 0x7FFFFFFFu) : ~key;
    float v = __uint_as_float(bits);
    if (!isfinite(v)) v = 0.f;
    ((float*)m_enc)[i] = v;
}

// ---------------- edge pass 2: exp + segment sum ----------------
__global__ void edge_sum(float* __restrict__ ew, const int* __restrict__ dst,
                         const float* __restrict__ m, float* __restrict__ den, int e_total) {
    int t = blockIdx.x * blockDim.x + threadIdx.x;
    int eid = t >> 3, head = t & 7;
    if (eid >= e_total) return;
    int d0 = dst[eid];
    float num = __expf(ew[t] - m[d0 * NHEAD + head]);
    ew[t] = num;
    atomicAdd(den + d0 * NHEAD + head, num);
}

// ---------------- edge pass 3: alpha-weighted scatter ----------------
__global__ void edge_scatter(const float* __restrict__ ew, const int* __restrict__ src,
                             const int* __restrict__ dst, const float* __restrict__ den,
                             const float* __restrict__ z, float* __restrict__ out, int e_total) {
    int t = blockIdx.x * blockDim.x + threadIdx.x;
    int eid = t >> 3, head = t & 7;
    if (eid >= e_total) return;
    int s = src[eid], d0 = dst[eid];
    float dn = den[d0 * NHEAD + head];
    float alpha = ew[t] / ((dn > 0.f) ? dn : 1.f);
    const float4* zr = (const float4*)(z + ((size_t)s * NHEAD + head) * DHEAD);
    float* orow = out + ((size_t)d0 * NHEAD + head) * DHEAD;
    #pragma unroll
    for (int q = 0; q < 8; ++q) {
        float4 zv = zr[q];
        atomicAdd(orow + q * 4 + 0, alpha * zv.x);
        atomicAdd(orow + q * 4 + 1, alpha * zv.y);
        atomicAdd(orow + q * 4 + 2, alpha * zv.z);
        atomicAdd(orow + q * 4 + 3, alpha * zv.w);
    }
}

// ---------------- launch ----------------
extern "C" void kernel_launch(void* const* d_in, const int* in_sizes, int n_in,
                              void* d_out, int out_size, void* d_ws, size_t ws_size,
                              hipStream_t stream) {
    (void)n_in; (void)out_size; (void)ws_size;
    const float* h      = (const float*)d_in[0];
    const float* W      = (const float*)d_in[1];
    const float* attn_w = (const float*)d_in[2];
    const int*   src    = (const int*)d_in[3];
    const int*   dst    = (const int*)d_in[4];
    float* out = (float*)d_out;

    const int n = in_sizes[0] / IN_DIM;    // 50000
    const int e = in_sizes[3];             // 800000

    char* ws = (char*)d_ws;
    size_t off = 0;
    _Float16* h16 = (_Float16*)(ws + off); off += (size_t)n * IN_DIM * sizeof(_Float16);
    _Float16* Bp  = (_Float16*)(ws + off); off += (size_t)NHEAD * 8 * 2 * 32 * 16 * sizeof(_Float16);
    float* z      = (float*)(ws + off);    off += (size_t)n * IN_DIM * sizeof(float);
    float* el     = (float*)(ws + off);    off += (size_t)n * NHEAD * sizeof(float);
    float* er     = (float*)(ws + off);    off += (size_t)n * NHEAD * sizeof(float);
    unsigned* m_enc = (unsigned*)(ws + off); off += (size_t)n * NHEAD * sizeof(unsigned);
    float* den    = (float*)(ws + off);    off += (size_t)n * NHEAD * sizeof(float);
    float* ew     = (float*)(ws + off);    off += (size_t)e * NHEAD * sizeof(float);

    const int TB = 256;
    int htotal = n * IN_DIM;
    conv_h_f16<<<(htotal + TB - 1) / TB, TB, 0, stream>>>(h, h16, htotal);
    pack_W_f16<<<(NHEAD * 8 * 2 * 32 * 16 + TB - 1) / TB, TB, 0, stream>>>(W, Bp);

    gemm_z<<<(n + 15) / 16, TB, 0, stream>>>(h16, Bp, z, n);

    int nh = n * NHEAD;
    attn_scores<<<(nh + TB - 1) / TB, TB, 0, stream>>>(z, attn_w, el, er, nh);
    init_bufs<<<(htotal + TB - 1) / TB, TB, 0, stream>>>(out, m_enc, den, htotal, nh);

    int eh = e * NHEAD;
    edge_max<<<(eh + TB - 1) / TB, TB, 0, stream>>>(el, er, src, dst, ew, m_enc, e);
    decode_m<<<(nh + TB - 1) / TB, TB, 0, stream>>>(m_enc, nh);
    edge_sum<<<(eh + TB - 1) / TB, TB, 0, stream>>>(ew, dst, (const float*)m_enc, den, e);
    edge_scatter<<<(eh + TB - 1) / TB, TB, 0, stream>>>(ew, src, dst, den, z, out, e);
}
